// SharedCodebook_75883482186300
// MI455X (gfx1250) — matrix operand order repaired and checked by reading, not compiled, not run on hardware
//
#include <hip/hip_runtime.h>

typedef __bf16 bf16;
typedef bf16  v16bf __attribute__((ext_vector_type(16)));
typedef float v8f   __attribute__((ext_vector_type(8)));

#define BATCH   65536
#define NCODES  8192
#define DIM     256
#define LSTRIDE 272       // bf16 elems per LDS row (256 + 16 pad, keeps 32B alignment)
#define TN_A    16        // codes per tile, async double-buffered main
#define NTILES  (NCODES / TN_A)
#define TN_C    32        // codes per tile, fallback convert-in-kernel main

__device__ __forceinline__ v8f wmma_bf16(v16bf a, v16bf b, v8f c) {
  return __builtin_amdgcn_wmma_f32_16x16x32_bf16(false, a, false, b, (short)0, c, false, false);
}

// CDNA5 async global->LDS copy, 16 bytes per lane, tracked by ASYNCcnt.
__device__ __forceinline__ void async_b128(const bf16* gsrc, bf16* ldst) {
  const unsigned lds_off = (unsigned)(size_t)ldst;           // generic LDS addr low 32 = LDS byte addr
  const unsigned long long ga = (unsigned long long)(size_t)gsrc;
  asm volatile("global_load_async_to_lds_b128 %0, %1, off"
               :: "v"(lds_off), "v"(ga)
               : "memory");
}

__device__ __forceinline__ void wait_async0() {
#if __has_builtin(__builtin_amdgcn_s_wait_asynccnt)
  __builtin_amdgcn_s_wait_asynccnt(0);
#else
  asm volatile("s_wait_asynccnt 0x0" ::: "memory");
#endif
}

// ---------------------------------------------------------------------------
// Kernel 1: codebook squared norms, zero histogram, optional bf16 hi/lo split
// ---------------------------------------------------------------------------
__global__ __launch_bounds__(256) void vq_prep(const float* __restrict__ codebook,
                                               float* __restrict__ c_sq,
                                               unsigned* __restrict__ hist,
                                               bf16* __restrict__ cb_hi,
                                               bf16* __restrict__ cb_lo,
                                               int write_cb) {
  __shared__ float red[256];
  const unsigned t = threadIdx.x;
  const size_t idx = (size_t)blockIdx.x * DIM + t;
  const float v = codebook[idx];
  if (write_cb) {
    const bf16 h = (bf16)v;
    cb_hi[idx] = h;
    cb_lo[idx] = (bf16)(v - (float)h);
  }
  red[t] = v * v;
  __syncthreads();
  for (unsigned s = 128; s > 0; s >>= 1) {
    if (t < s) red[t] += red[t + s];
    __syncthreads();
  }
  if (t == 0) {
    c_sq[blockIdx.x] = red[0];
    hist[blockIdx.x] = 0u;
  }
}

// ---------------------------------------------------------------------------
// Shared device helpers for both mains
// ---------------------------------------------------------------------------
__device__ __forceinline__ void load_a_frags(const float* __restrict__ features,
                                             unsigned m0w, unsigned l15, unsigned hf,
                                             v16bf* a_hi, v16bf* a_lo) {
  // A 16x32 layout (contiguous, matches B): lanes 0-15 K=[0,16); lanes 16-31 K=[16,32)
  const float* fr = features + (size_t)(m0w + l15) * DIM;
#pragma unroll
  for (int kc = 0; kc < 8; ++kc) {
    const unsigned k1 = (unsigned)kc * 32u + hf * 16u;
    const unsigned k2 = k1 + 8u;
#pragma unroll
    for (int j = 0; j < 8; ++j) {
      const float x = fr[k1 + j];
      const bf16 h = (bf16)x;
      a_hi[kc][j] = h;
      a_lo[kc][j] = (bf16)(x - (float)h);
      const float y = fr[k2 + j];
      const bf16 h2 = (bf16)y;
      a_hi[kc][8 + j] = h2;
      a_lo[kc][8 + j] = (bf16)(y - (float)h2);
    }
  }
}

// Argmin reduce + histogram + gather + STE output + deterministic loss partial.
__device__ __forceinline__ void epilogue(const float* __restrict__ features,
                                         const float* __restrict__ codebook,
                                         unsigned* __restrict__ hist,
                                         float* __restrict__ partial,
                                         float* __restrict__ out,
                                         float* bestd, unsigned* besti,
                                         unsigned* sBest, float* sLoss,
                                         unsigned tid, unsigned lane, unsigned wv,
                                         unsigned hf, unsigned l15, unsigned m0w) {
#pragma unroll
  for (int r = 0; r < 8; ++r) {
    float bm = bestd[r];
    unsigned bi = besti[r];
#pragma unroll
    for (int off = 8; off >= 1; off >>= 1) {
      const float    om = __shfl_xor(bm, off, 32);
      const unsigned oi = __shfl_xor(bi, off, 32);
      if (om < bm || (om == bm && oi < bi)) { bm = om; bi = oi; }  // first-index tiebreak
    }
    if (l15 == 0u) {                       // lanes 0 (rows r) and 16 (rows r+8)
      sBest[wv * 16u + hf * 8u + (unsigned)r] = bi;
      atomicAdd(&hist[bi], 1u);
    }
  }
  __syncthreads();

  float lsum = 0.0f;
#pragma unroll 1
  for (int m = 0; m < 16; ++m) {
    const unsigned c = sBest[wv * 16u + (unsigned)m];
    const float4* q4 = (const float4*)(codebook + (size_t)c * DIM);
    const float4* f4 = (const float4*)(features + (size_t)(m0w + (unsigned)m) * DIM);
    float4* o4 = (float4*)(out + (size_t)(m0w + (unsigned)m) * DIM);
#pragma unroll
    for (int j = 0; j < 2; ++j) {
      const float4 q = q4[lane + j * 32];
      const float4 f = f4[lane + j * 32];
      const float dx = q.x - f.x, dy = q.y - f.y, dz = q.z - f.z, dw = q.w - f.w;
      float4 o;
      o.x = f.x + dx; o.y = f.y + dy; o.z = f.z + dz; o.w = f.w + dw;   // exact STE form
      o4[lane + j * 32] = o;
      lsum += dx * dx + dy * dy + dz * dz + dw * dw;
    }
  }
#pragma unroll
  for (int off = 16; off >= 1; off >>= 1) lsum += __shfl_xor(lsum, off, 32);
  if (lane == 0) sLoss[wv] = lsum;
  __syncthreads();
  if (tid == 0) {
    float s = 0.0f;
#pragma unroll
    for (int w = 0; w < 8; ++w) s += sLoss[w];
    partial[blockIdx.x] = s;   // deterministic: plain store, fixed order
  }
}

// ---------------------------------------------------------------------------
// Kernel 2a (preferred): async double-buffered WMMA distance GEMM + argmin
// ---------------------------------------------------------------------------
__global__ __launch_bounds__(256) void vq_main_async(const float* __restrict__ features,
                                                     const float* __restrict__ codebook,
                                                     const bf16* __restrict__ cb_hi,
                                                     const bf16* __restrict__ cb_lo,
                                                     const float* __restrict__ c_sq,
                                                     unsigned* __restrict__ hist,
                                                     float* __restrict__ partial,
                                                     float* __restrict__ out) {
  __shared__ __align__(32) bf16 sBh[2][TN_A * LSTRIDE];
  __shared__ __align__(32) bf16 sBl[2][TN_A * LSTRIDE];
  __shared__ unsigned sBest[8 * 16];
  __shared__ float sLoss[8];

  const unsigned tid = threadIdx.x;
  const unsigned lane = tid & 31u;
  const unsigned wv  = tid >> 5;
  const unsigned hf  = lane >> 4;
  const unsigned l15 = lane & 15u;
  const unsigned m0w = (blockIdx.x * 8u + wv) * 16u;

  v16bf a_hi[8], a_lo[8];
  load_a_frags(features, m0w, l15, hf, a_hi, a_lo);

  // 16 rows x 256 bf16 per array = 8KB = 512 x 16B chunks -> 2 per thread
  auto load_tile = [&](unsigned buf, unsigned cbase) {
#pragma unroll
    for (int q = 0; q < 2; ++q) {
      const unsigned chunk = tid + (unsigned)q * 256u;   // 0..511
      const unsigned row = chunk >> 5;                   // 32 chunks per row
      const unsigned off = (chunk & 31u) * 8u;           // bf16 elems (16B granules)
      const size_t gidx = (size_t)(cbase + row) * DIM + off;
      const unsigned lidx = row * LSTRIDE + off;
      async_b128(cb_hi + gidx, &sBh[buf][lidx]);
      async_b128(cb_lo + gidx, &sBl[buf][lidx]);
    }
  };

  load_tile(0u, 0u);
  wait_async0();
  __syncthreads();

  float    bestd[8];
  unsigned besti[8];
#pragma unroll
  for (int r = 0; r < 8; ++r) { bestd[r] = 3.4e38f; besti[r] = 0u; }

  for (unsigned i = 0; i < NTILES; ++i) {
    const unsigned cbase = i * TN_A;
    const unsigned buf = i & 1u;
    if (i + 1u < NTILES) load_tile(buf ^ 1u, cbase + TN_A);   // overlap with WMMAs below

    const unsigned code = cbase + l15;
    const float csq = c_sq[code];
    const bf16* bh = &sBh[buf][l15 * LSTRIDE + hf * 16u];
    const bf16* bl = &sBl[buf][l15 * LSTRIDE + hf * 16u];
    v8f acc0 = {}, acc1 = {};                 // two independent chains for ILP
#pragma unroll
    for (int kc = 0; kc < 4; ++kc) {
      const v16bf Bh0 = *(const v16bf*)(bh + kc * 32);
      const v16bf Bl0 = *(const v16bf*)(bl + kc * 32);
      const v16bf Bh1 = *(const v16bf*)(bh + (kc + 4) * 32);
      const v16bf Bl1 = *(const v16bf*)(bl + (kc + 4) * 32);
      acc0 = wmma_bf16(a_hi[kc],     Bh0, acc0);
      acc1 = wmma_bf16(a_hi[kc + 4], Bh1, acc1);
      acc0 = wmma_bf16(a_lo[kc],     Bh0, acc0);
      acc1 = wmma_bf16(a_lo[kc + 4], Bh1, acc1);
      acc0 = wmma_bf16(a_hi[kc],     Bl0, acc0);
      acc1 = wmma_bf16(a_hi[kc + 4], Bl1, acc1);
    }
    const v8f acc = acc0 + acc1;
#pragma unroll
    for (int r = 0; r < 8; ++r) {
      const float dist = csq - 2.0f * acc[r];
      if (dist < bestd[r]) { bestd[r] = dist; besti[r] = code; }
    }
    wait_async0();      // next tile landed (hidden behind the 24 WMMAs above)
    __syncthreads();
  }

  epilogue(features, codebook, hist, partial, out,
           bestd, besti, sBest, sLoss, tid, lane, wv, hf, l15, m0w);
}

// ---------------------------------------------------------------------------
// Kernel 2b (fallback if ws too small): convert-in-kernel tiles
// ---------------------------------------------------------------------------
__global__ __launch_bounds__(256) void vq_main_conv(const float* __restrict__ features,
                                                    const float* __restrict__ codebook,
                                                    const float* __restrict__ c_sq,
                                                    unsigned* __restrict__ hist,
                                                    float* __restrict__ partial,
                                                    float* __restrict__ out) {
  __shared__ __align__(32) bf16 sBh[TN_C * LSTRIDE];
  __shared__ __align__(32) bf16 sBl[TN_C * LSTRIDE];
  __shared__ unsigned sBest[8 * 16];
  __shared__ float sLoss[8];

  const unsigned tid = threadIdx.x;
  const unsigned lane = tid & 31u;
  const unsigned wv  = tid >> 5;
  const unsigned hf  = lane >> 4;
  const unsigned l15 = lane & 15u;
  const unsigned m0w = (blockIdx.x * 8u + wv) * 16u;

  v16bf a_hi[8], a_lo[8];
  load_a_frags(features, m0w, l15, hf, a_hi, a_lo);

  float    bestd[8];
  unsigned besti[8];
#pragma unroll
  for (int r = 0; r < 8; ++r) { bestd[r] = 3.4e38f; besti[r] = 0u; }

  for (unsigned cbase = 0; cbase < NCODES; cbase += TN_C) {
    __syncthreads();
    for (unsigned i = tid; i < TN_C * (DIM / 4); i += 256u) {
      const unsigned row = i >> 6;
      const unsigned col = (i & 63u) << 2;
      const float4 v = *(const float4*)(codebook + (size_t)(cbase + row) * DIM + col);
      bf16* dh = sBh + row * LSTRIDE + col;
      bf16* dl = sBl + row * LSTRIDE + col;
      const bf16 h0 = (bf16)v.x, h1 = (bf16)v.y, h2 = (bf16)v.z, h3 = (bf16)v.w;
      dh[0] = h0; dh[1] = h1; dh[2] = h2; dh[3] = h3;
      dl[0] = (bf16)(v.x - (float)h0);
      dl[1] = (bf16)(v.y - (float)h1);
      dl[2] = (bf16)(v.z - (float)h2);
      dl[3] = (bf16)(v.w - (float)h3);
    }
    __syncthreads();

    for (unsigned nt = 0; nt < TN_C / 16u; ++nt) {
      const unsigned n = nt * 16u + l15;
      const unsigned code = cbase + n;
      const float csq = c_sq[code];
      const bf16* bh = sBh + n * LSTRIDE + hf * 16u;
      const bf16* bl = sBl + n * LSTRIDE + hf * 16u;
      v8f acc = {};
#pragma unroll
      for (int kc = 0; kc < 8; ++kc) {
        const v16bf Bh = *(const v16bf*)(bh + kc * 32);
        const v16bf Bl = *(const v16bf*)(bl + kc * 32);
        acc = wmma_bf16(a_hi[kc], Bh, acc);
        acc = wmma_bf16(a_lo[kc], Bh, acc);
        acc = wmma_bf16(a_hi[kc], Bl, acc);
      }
#pragma unroll
      for (int r = 0; r < 8; ++r) {
        const float dist = csq - 2.0f * acc[r];
        if (dist < bestd[r]) { bestd[r] = dist; besti[r] = code; }
      }
    }
  }

  epilogue(features, codebook, hist, partial, out,
           bestd, besti, sBest, sLoss, tid, lane, wv, hf, l15, m0w);
}

// ---------------------------------------------------------------------------
// Kernel 3: entropy over histogram + final loss scalars (fixed-order reduce)
// ---------------------------------------------------------------------------
__global__ __launch_bounds__(256) void vq_finalize(const unsigned* __restrict__ hist,
                                                   const float* __restrict__ partial,
                                                   float* __restrict__ out_scalars) {
  __shared__ float rE[256];
  __shared__ float rL[256];
  const unsigned t = threadIdx.x;
  float e = 0.0f, l = 0.0f;
  for (unsigned c = t; c < NCODES; c += 256u) {
    const float p = (float)hist[c] * (1.0f / (float)BATCH);
    e -= p * logf(p + 1e-8f);
  }
  for (unsigned i = t; i < 512u; i += 256u) l += partial[i];
  rE[t] = e; rL[t] = l;
  __syncthreads();
  for (unsigned s = 128; s > 0; s >>= 1) {
    if (t < s) { rE[t] += rE[t + s]; rL[t] += rL[t + s]; }
    __syncthreads();
  }
  if (t == 0) {
    const float lm = rL[0] * (1.0f / ((float)BATCH * (float)DIM));
    out_scalars[0] = lm;   // codebook_loss
    out_scalars[1] = lm;   // commitment_loss (numerically identical)
    out_scalars[2] = rE[0];
  }
}

// ---------------------------------------------------------------------------
extern "C" void kernel_launch(void* const* d_in, const int* in_sizes, int n_in,
                              void* d_out, int out_size, void* d_ws, size_t ws_size,
                              hipStream_t stream) {
  (void)in_sizes; (void)n_in; (void)out_size;
  const float* features = (const float*)d_in[0];   // [65536, 256] f32
  const float* codebook = (const float*)d_in[1];   // [8192, 256] f32
  float* out = (float*)d_out;                      // [65536*256] + 3 scalars

  char* ws = (char*)d_ws;
  float*    c_sq    = (float*)ws;                        // 8192 f32  (32 KB)
  unsigned* hist    = (unsigned*)(ws + 32768);           // 8192 u32  (32 KB)
  float*    partial = (float*)(ws + 65536);              // 512 f32   (2 KB)
  bf16*     cb_hi   = (bf16*)(ws + 131072);              // 8192*256 bf16 (4 MB)
  bf16*     cb_lo   = (bf16*)(ws + 131072 + (size_t)NCODES * DIM * 2);

  const size_t ws_need = 131072 + 2 * (size_t)NCODES * DIM * sizeof(bf16);
  const int use_async = (ws_size >= ws_need);

  vq_prep<<<NCODES, 256, 0, stream>>>(codebook, c_sq, hist, cb_hi, cb_lo, use_async);
  if (use_async) {
    vq_main_async<<<BATCH / 128, 256, 0, stream>>>(features, codebook, cb_hi, cb_lo,
                                                   c_sq, hist, partial, out);
  } else {
    vq_main_conv<<<BATCH / 128, 256, 0, stream>>>(features, codebook,
                                                  c_sq, hist, partial, out);
  }
  vq_finalize<<<1, 256, 0, stream>>>(hist, partial, out + (size_t)BATCH * DIM);
}
